// LSTM_8847632630344
// MI455X (gfx1250) — matrix operand order repaired and checked by reading, not compiled, not run on hardware
//
#include <hip/hip_runtime.h>

#define B_SZ  256
#define K_T   512
#define I_DIM 64
#define H_DIM 512
#define G4    2048

typedef __attribute__((ext_vector_type(16))) __bf16 v16bf;
typedef __attribute__((ext_vector_type(8)))  __bf16 v8bf;
typedef __attribute__((ext_vector_type(8)))  float  v8f;

union ABf { v8bf h[2]; v16bf v; };

// sigmoid(x) = 1/(1+e^-x) using v_exp_f32 + v_rcp_f32 (TRANS32, co-executes with VALU)
__device__ __forceinline__ float fast_sigmoid(float x) {
    return __builtin_amdgcn_rcpf(1.0f + __expf(-x));
}
// tanh(x) = 2*sigmoid(2x) - 1
__device__ __forceinline__ float fast_tanh(float x) {
    return 2.0f * fast_sigmoid(2.0f * x) - 1.0f;
}

// One-time prep: convert R (4H,H) to bf16 (row-major == column-major of B=R^T),
// transpose-convert kernel (I,4H) -> Wkt (4H,I) bf16, zero h0 and c.
__global__ void lstm_prep(const float* __restrict__ kernel_w,
                          const float* __restrict__ rkernel,
                          __bf16* __restrict__ Rbf,
                          __bf16* __restrict__ Wkt,
                          __bf16* __restrict__ h0,
                          float*  __restrict__ c)
{
    const int NR = G4 * H_DIM;   // 1048576
    const int NW = G4 * I_DIM;   // 131072
    const int NH = B_SZ * H_DIM; // 131072
    const int total = NR + NW + NH + NH;
    for (int idx = blockIdx.x * blockDim.x + threadIdx.x; idx < total;
         idx += gridDim.x * blockDim.x) {
        if (idx < NR) {
            Rbf[idx] = (__bf16)rkernel[idx];
        } else if (idx < NR + NW) {
            int j = idx - NR;
            int col = j >> 6;          // 4H index
            int i   = j & 63;          // I index
            Wkt[j] = (__bf16)kernel_w[i * G4 + col];
        } else if (idx < NR + NW + NH) {
            h0[idx - NR - NW] = (__bf16)0.0f;
        } else {
            c[idx - NR - NW - NH] = 0.0f;
        }
    }
}

// One LSTM timestep: gates = x_t@Wk + h@R^T + bias, fused state update.
// Block = 128 threads (4 waves, 2x2). Tile: 32 batch x 32 hidden, all 4 gates.
__global__ __launch_bounds__(128) void lstm_step(
    const float*  __restrict__ x,       // input_seq (B,K,I) f32
    const float*  __restrict__ bias,    // (4H)
    const __bf16* __restrict__ Wkt,     // (4H, I) bf16, K-contiguous per column
    const __bf16* __restrict__ Rbf,     // (4H, H) bf16, B[k][n] = Rbf[n*H + k]
    const __bf16* __restrict__ h_read,  // (B, H) bf16
    __bf16*       __restrict__ h_write, // (B, H) bf16
    float*        __restrict__ c,       // (B, H) f32
    int t)
{
    const int lane = threadIdx.x & 31;
    const int wave = threadIdx.x >> 5;
    const int wm = wave >> 1, wn = wave & 1;
    const int lrow = lane & 15;     // M for A, N for B/C/D
    const int lhi  = lane >> 4;     // lane-half: K/M offset selector

    const int bm = blockIdx.x * 32 + wm * 16;   // batch tile origin
    const int n0 = blockIdx.y * 32 + wn * 16;   // hidden-col tile origin

    // Bias for this lane's output column (D: N = lane%16), folded into C operand.
    const int hcol = n0 + lrow;
    const float bF = bias[hcol];
    const float bI = bias[H_DIM + hcol];
    const float bC = bias[2 * H_DIM + hcol];
    const float bO = bias[3 * H_DIM + hcol];

    v8f acc[4];
    #pragma unroll
    for (int r = 0; r < 8; ++r) {
        acc[0][r] = bF; acc[1][r] = bI; acc[2][r] = bC; acc[3][r] = bO;
    }

    const int m  = bm + lrow;                   // A row (batch) for this lane
    const int nb = n0 + lrow;                   // B col (hidden) for this lane

    // ---- recurrent term: K over H_DIM, h in bf16 ----
    const __bf16* aRow = h_read + (size_t)m * H_DIM;
    #pragma unroll 2
    for (int kc = 0; kc < H_DIM / 32; ++kc) {
        ABf a;
        const int ka = kc * 32 + lhi * 16;      // A: one 16-elem contiguous run
        a.h[0] = *(const v8bf*)(aRow + ka);
        a.h[1] = *(const v8bf*)(aRow + ka + 8);
        const int kb = kc * 32 + lhi * 16;      // B: one 16-elem contiguous run
        #pragma unroll
        for (int g = 0; g < 4; ++g) {
            const __bf16* bp = Rbf + ((size_t)(g * H_DIM + nb)) * H_DIM + kb;
            v16bf bfrag = *(const v16bf*)bp;
            acc[g] = __builtin_amdgcn_wmma_f32_16x16x32_bf16(
                false, a.v, false, bfrag, (short)0, acc[g], false, false);
        }
    }

    // ---- input-projection term: K over I_DIM, x converted f32->bf16 on load ----
    const float* xRow = x + ((size_t)m * K_T + t) * I_DIM;
    #pragma unroll
    for (int kc = 0; kc < I_DIM / 32; ++kc) {
        ABf a;
        const int ka = kc * 32 + lhi * 16;
        #pragma unroll
        for (int j = 0; j < 8; ++j) a.v[j]     = (__bf16)xRow[ka + j];
        #pragma unroll
        for (int j = 0; j < 8; ++j) a.v[8 + j] = (__bf16)xRow[ka + 8 + j];
        const int kb = kc * 32 + lhi * 16;
        #pragma unroll
        for (int g = 0; g < 4; ++g) {
            const __bf16* bp = Wkt + ((size_t)(g * H_DIM + nb)) * I_DIM + kb;
            v16bf bfrag = *(const v16bf*)bp;
            acc[g] = __builtin_amdgcn_wmma_f32_16x16x32_bf16(
                false, a.v, false, bfrag, (short)0, acc[g], false, false);
        }
    }

    // ---- fused gate nonlinearities + cell/hidden update (bias already in acc) ----
    #pragma unroll
    for (int r = 0; r < 8; ++r) {
        const int b_idx = bm + lhi * 8 + r;     // D: M = vgpr + 8*lane-half
        const size_t idx = (size_t)b_idx * H_DIM + hcol;
        float f  = fast_sigmoid(acc[0][r]);
        float ig = fast_sigmoid(acc[1][r]);
        float cg = fast_tanh(acc[2][r]);
        float og = fast_sigmoid(acc[3][r]);
        float cn = f * c[idx] + ig * cg;
        c[idx] = cn;
        h_write[idx] = (__bf16)(og * fast_tanh(cn));
    }
}

// out[b] = h_last[b,:] . Wout + bout
__global__ void lstm_out(const __bf16* __restrict__ hfin,
                         const float*  __restrict__ Wout,
                         const float*  __restrict__ bout,
                         float*        __restrict__ out)
{
    int b = blockIdx.x * blockDim.x + threadIdx.x;
    if (b >= B_SZ) return;
    float accv = bout[0];
    const __bf16* hp = hfin + (size_t)b * H_DIM;
    #pragma unroll 8
    for (int h = 0; h < H_DIM; ++h)
        accv += (float)hp[h] * Wout[h];
    out[b] = accv;
}

extern "C" void kernel_launch(void* const* d_in, const int* in_sizes, int n_in,
                              void* d_out, int out_size, void* d_ws, size_t ws_size,
                              hipStream_t stream) {
    const float* input_seq = (const float*)d_in[0];
    const float* kernel_w  = (const float*)d_in[1];
    const float* rkernel   = (const float*)d_in[2];
    const float* bias      = (const float*)d_in[3];
    const float* Wout      = (const float*)d_in[4];
    const float* bout      = (const float*)d_in[5];
    float* out = (float*)d_out;

    char* ws = (char*)d_ws;
    __bf16* Rbf = (__bf16*)(ws);                            // 2 MB
    __bf16* Wkt = (__bf16*)(ws + 2097152);                  // 256 KB
    __bf16* h0  = (__bf16*)(ws + 2097152 + 262144);         // 256 KB
    __bf16* h1  = (__bf16*)(ws + 2097152 + 2 * 262144);     // 256 KB
    float*  c   = (float*) (ws + 2097152 + 3 * 262144);     // 512 KB

    lstm_prep<<<1024, 256, 0, stream>>>(kernel_w, rkernel, Rbf, Wkt, h0, c);

    dim3 grid(B_SZ / 32, H_DIM / 32);   // 8 x 16 = 128 workgroups
    for (int t = 0; t < K_T; ++t) {
        const __bf16* hr = (t & 1) ? h1 : h0;
        __bf16*       hw = (t & 1) ? h0 : h1;
        lstm_step<<<grid, 128, 0, stream>>>(input_seq, bias, Wkt, Rbf, hr, hw, c, t);
    }
    // T=512 (even #steps): final h lands in h0
    lstm_out<<<1, 256, 0, stream>>>(h0, Wout, bout, out);
}
